// SelfMultiheadAttn_14370960572998
// MI455X (gfx1250) — matrix operand order repaired and checked by reading, not compiled, not run on hardware
//
#include <hip/hip_runtime.h>
#include <cstddef>

// Problem constants (from reference): T=2048, B=4, E=1024, H=16, D=64
#define TDIM 2048
#define BDIM 4
#define EDIM 1024
#define HDIM 16
#define DDIM 64
#define MDIM (TDIM * BDIM)          // 8192 flattened (t,b) rows
#define QK_SCALE 0.125f             // 64^-0.5

typedef __attribute__((ext_vector_type(16))) __bf16        v16bf;
typedef __attribute__((ext_vector_type(8)))  float         v8f;
typedef __attribute__((ext_vector_type(4)))  unsigned int  u32x4;
typedef __attribute__((ext_vector_type(4)))  float         f32x4;

// bf16 WMMA fragment viewed as raw u16 lanes / 128-bit load chunks.
union BF16Frag {
    v16bf          v;
    unsigned short u[16];
    u32x4          q[2];
};

__device__ __forceinline__ unsigned short f2bf(float f) {
    unsigned int u = __float_as_uint(f);
    u += 0x7FFFu + ((u >> 16) & 1u);   // round-to-nearest-even
    return (unsigned short)(u >> 16);
}

__device__ __forceinline__ v8f wmma_bf16(v16bf a, v16bf b, v8f c) {
    // D(f32 16x16) = A(bf16 16x32) * B(bf16 32x16) + C
    return __builtin_amdgcn_wmma_f32_16x16x32_bf16(false, a, false, b, (short)0, c, false, false);
}

// ---------------------------------------------------------------------------
// Kernel 0: one-shot fp32 -> bf16 conversion (x and the 4 weight matrices).
// Hoists ALL conversion VALU work out of the GEMM hot loops; each source
// element is converted exactly once instead of once per consuming tile.
// 8 elements per thread, b128 in / b128 out. n must be a multiple of 8.
// ---------------------------------------------------------------------------
__global__ void __launch_bounds__(256) cvt_f32_to_bf16(
    const float* __restrict__ src, unsigned short* __restrict__ dst)
{
    const size_t i = ((size_t)blockIdx.x * 256 + threadIdx.x) * 8;
    f32x4 a = *(const f32x4*)(src + i);
    f32x4 b = *(const f32x4*)(src + i + 4);
    union { unsigned short u[8]; u32x4 q; } o;
#pragma unroll
    for (int j = 0; j < 4; ++j) { o.u[j] = f2bf(a[j]); o.u[4 + j] = f2bf(b[j]); }
    *(u32x4*)(dst + i) = o.q;
}

// ---------------------------------------------------------------------------
// Kernel 1: fused Q/K/V projection GEMM, all-bf16 operands.
//   mat 0 -> Q (scaled by QK_SCALE), [T,B,E] row-major
//   mat 1 -> K, [T,B,E] row-major
//   mat 2 -> V stored TRANSPOSED as vT[B,H,D,T] for contiguous P@V fragments
// Each wave: 32x32 tile; per K=32 step: 8x global_load_b128 + 4x WMMA, no
// per-element VALU.
// ---------------------------------------------------------------------------
__global__ void __launch_bounds__(256) qkv_gemm_bf16(
    const unsigned short* __restrict__ xb,
    const unsigned short* __restrict__ wq, const unsigned short* __restrict__ wk,
    const unsigned short* __restrict__ wv,
    const float* __restrict__ qbias, const float* __restrict__ kbias, const float* __restrict__ vbias,
    unsigned short* __restrict__ qo, unsigned short* __restrict__ ko, unsigned short* __restrict__ vTo)
{
    const int lane = threadIdx.x & 31;
    const int wave = blockIdx.x * 8 + (threadIdx.x >> 5);
    const int mat  = wave >> 13;          // 8192 tiles per matrix
    const int tid  = wave & 8191;
    const int m0   = (tid >> 5) << 5;     // 256 M-tiles of 32
    const int n0   = (tid & 31) << 5;     // 32  N-tiles of 32
    const int l16  = lane & 15;
    const int half = lane >> 4;

    const unsigned short* W    = (mat == 0) ? wq : (mat == 1) ? wk : wv;
    const float*          bias = (mat == 0) ? qbias : (mat == 1) ? kbias : vbias;

    v8f acc00 = {}, acc01 = {}, acc10 = {}, acc11 = {};

    // ISA A-frag (16-bit 16x32): lanes<16 -> K {0..7,16..23}; lanes>=16 -> K {8..15,24..31}
    const unsigned short* arow0 = xb + (size_t)(m0 + l16) * EDIM + half * 8;
    const unsigned short* arow1 = arow0 + (size_t)16 * EDIM;
    // ISA B-frag (16-bit 32x16): lanes<16 -> K {0..7,16..23}; lanes>=16 -> K {8..15,24..31}; col = lane%16
    const unsigned short* brow0 = W + (size_t)(n0 + l16) * EDIM + half * 8;
    const unsigned short* brow1 = brow0 + (size_t)16 * EDIM;

    for (int k0 = 0; k0 < EDIM; k0 += 32) {
        __builtin_prefetch(brow0 + k0 + 128, 0, 1);
        BF16Frag a0, a1, b0, b1;
        a0.q[0] = *(const u32x4*)(arow0 + k0);
        a0.q[1] = *(const u32x4*)(arow0 + k0 + 16);
        a1.q[0] = *(const u32x4*)(arow1 + k0);
        a1.q[1] = *(const u32x4*)(arow1 + k0 + 16);
        b0.q[0] = *(const u32x4*)(brow0 + k0);
        b0.q[1] = *(const u32x4*)(brow0 + k0 + 16);
        b1.q[0] = *(const u32x4*)(brow1 + k0);
        b1.q[1] = *(const u32x4*)(brow1 + k0 + 16);
        acc00 = wmma_bf16(a0.v, b0.v, acc00);
        acc01 = wmma_bf16(a0.v, b1.v, acc01);
        acc10 = wmma_bf16(a1.v, b0.v, acc10);
        acc11 = wmma_bf16(a1.v, b1.v, acc11);
    }

    const float scale = (mat == 0) ? QK_SCALE : 1.0f;
#pragma unroll
    for (int tile = 0; tile < 4; ++tile) {
        const v8f& c = (tile == 0) ? acc00 : (tile == 1) ? acc01 : (tile == 2) ? acc10 : acc11;
        const int mb = (tile >> 1) << 4;
        const int nb = (tile & 1) << 4;
#pragma unroll
        for (int r = 0; r < 8; ++r) {
            // C f32 layout: VGPR r -> row r + 8*half, col = lane%16
            const int row = m0 + mb + r + half * 8;   // row = t*4 + b
            const int col = n0 + nb + l16;            // col = h*64 + d
            const unsigned short hv = f2bf((c[r] + bias[col]) * scale);
            if (mat == 2) {
                const int t = row >> 2, b = row & 3;
                const int h = col >> 6, d = col & 63;
                vTo[(((size_t)((b << 4) + h) << 6) + (size_t)d) * TDIM + t] = hv;
            } else {
                unsigned short* dst = (mat == 0) ? qo : ko;
                dst[(size_t)row * EDIM + col] = hv;
            }
        }
    }
}

// ---------------------------------------------------------------------------
// Kernel 2: flash attention. One wave per (b, h, 16-query tile).
// Per 32-key iteration: 4 WMMAs for S = q*k^T, online softmax, LDS round-trip
// to reshape P (C layout -> A layout), 4 WMMAs for acc += P @ vT.
// ---------------------------------------------------------------------------
__global__ void __launch_bounds__(256) attn_flash_bf16(
    const unsigned short* __restrict__ qb,
    const unsigned short* __restrict__ kb,
    const unsigned short* __restrict__ vT,
    unsigned short* __restrict__ ctx)
{
    __shared__ unsigned short plds[8 * 16 * 40];   // 8 waves x 16 rows x 40 u16 (80B stride, 16B aligned)
    const int lane  = threadIdx.x & 31;
    const int wslot = threadIdx.x >> 5;
    unsigned short* my = plds + wslot * 640;

    const int wave = blockIdx.x * 8 + wslot;
    const int bh   = wave >> 7;            // 128 query tiles per (b,h)
    const int t0   = (wave & 127) << 4;
    const int b    = bh >> 4;
    const int h    = bh & 15;
    const int l16  = lane & 15;
    const int half = lane >> 4;

    // Q A-fragments (16 rows x 64 d -> two K=32 fragments), loaded once.
    BF16Frag qa[2];
    {
        const int t = t0 + l16;
        const unsigned short* qp = qb + (size_t)(t * BDIM + b) * EDIM + h * DDIM;
#pragma unroll
        for (int dd = 0; dd < 2; ++dd) {
            const unsigned short* p = qp + dd * 32 + half * 8;
            qa[dd].q[0] = *(const u32x4*)(p);
            qa[dd].q[1] = *(const u32x4*)(p + 16);
        }
    }

    float mrun[8], lrun[8];
    v8f acc[4];
#pragma unroll
    for (int r = 0; r < 8; ++r) { mrun[r] = -3.0e38f; lrun[r] = 0.0f; }
#pragma unroll
    for (int nd = 0; nd < 4; ++nd)
#pragma unroll
        for (int r = 0; r < 8; ++r) acc[nd][r] = 0.0f;

    for (int s0 = 0; s0 < TDIM; s0 += 32) {
        // ---- S = q @ k^T for 32 keys (two 16-key C tiles) ----
        v8f S0 = {}, S1 = {};
#pragma unroll
        for (int sb = 0; sb < 2; ++sb) {
            const int s = s0 + sb * 16 + l16;   // B-frag col = key index
            const unsigned short* kp = kb + (size_t)(s * BDIM + b) * EDIM + h * DDIM;
            BF16Frag kfa, kfb;                  // d 0..31 and d 32..63
            { const unsigned short* p = kp + half * 8;      kfa.q[0] = *(const u32x4*)p; kfa.q[1] = *(const u32x4*)(p + 16); }
            { const unsigned short* p = kp + 32 + half * 8; kfb.q[0] = *(const u32x4*)p; kfb.q[1] = *(const u32x4*)(p + 16); }
            if (sb == 0) {
                S0 = wmma_bf16(qa[0].v, kfa.v, S0);
                S0 = wmma_bf16(qa[1].v, kfb.v, S0);
            } else {
                S1 = wmma_bf16(qa[0].v, kfa.v, S1);
                S1 = wmma_bf16(qa[1].v, kfb.v, S1);
            }
        }

        // ---- online softmax over these 32 columns ----
        float corr[8];
#pragma unroll
        for (int r = 0; r < 8; ++r) {
            float mc = fmaxf(S0[r], S1[r]);
#pragma unroll
            for (int off = 1; off < 16; off <<= 1)      // stays within 16-lane half (row owner group)
                mc = fmaxf(mc, __shfl_xor(mc, off, 32));
            const float mn = fmaxf(mrun[r], mc);
            const float cf = __expf(mrun[r] - mn);
            corr[r] = cf;
            mrun[r] = mn;
            const float p0 = __expf(S0[r] - mn);
            const float p1 = __expf(S1[r] - mn);
            S0[r] = p0; S1[r] = p1;
            float sum = p0 + p1;
#pragma unroll
            for (int off = 1; off < 16; off <<= 1)
                sum += __shfl_xor(sum, off, 32);
            lrun[r] = lrun[r] * cf + sum;
        }
#pragma unroll
        for (int nd = 0; nd < 4; ++nd)
#pragma unroll
            for (int r = 0; r < 8; ++r) acc[nd][r] *= corr[r];

        // ---- P: C layout -> LDS (16x32 bf16 tile) -> A-frag layout ----
#pragma unroll
        for (int r = 0; r < 8; ++r) {
            const int row = r + half * 8;
            my[row * 40 + l16]      = f2bf(S0[r]);
            my[row * 40 + 16 + l16] = f2bf(S1[r]);
        }
        BF16Frag pf;
        {
            const unsigned short* p = my + l16 * 40 + half * 8;  // A-frag row = lane%16
            pf.q[0] = *(const u32x4*)(p);        // K {0..7} or {8..15}
            pf.q[1] = *(const u32x4*)(p + 16);   // K {16..23} or {24..31}
        }

        // ---- acc += P(16x32) @ vT-fragments(32 keys x 16 d-cols) ----
#pragma unroll
        for (int nd = 0; nd < 4; ++nd) {
            const int d = nd * 16 + l16;
            const unsigned short* p = vT + (((size_t)bh << 6) + (size_t)d) * TDIM + s0 + half * 8;
            BF16Frag vf;
            vf.q[0] = *(const u32x4*)(p);
            vf.q[1] = *(const u32x4*)(p + 16);
            acc[nd] = wmma_bf16(pf.v, vf.v, acc[nd]);
        }
    }

    // ---- normalize and store ctx as bf16 [T,B,E] ----
#pragma unroll
    for (int r = 0; r < 8; ++r) {
        const float inv = 1.0f / lrun[r];
        const int t = t0 + r + half * 8;
        unsigned short* op = ctx + (size_t)(t * BDIM + b) * EDIM + h * DDIM;
#pragma unroll
        for (int nd = 0; nd < 4; ++nd)
            op[nd * 16 + l16] = f2bf(acc[nd][r] * inv);
    }
}

// ---------------------------------------------------------------------------
// Kernel 3: output projection. out = ctx(bf16) @ Wout^T(bf16) + bias, fp32 out.
// ---------------------------------------------------------------------------
__global__ void __launch_bounds__(256) out_proj_bf16(
    const unsigned short* __restrict__ ctx,
    const unsigned short* __restrict__ W,
    const float* __restrict__ bias,
    float* __restrict__ out)
{
    const int lane = threadIdx.x & 31;
    const int wave = blockIdx.x * 8 + (threadIdx.x >> 5);
    const int m0   = (wave >> 5) << 5;
    const int n0   = (wave & 31) << 5;
    const int l16  = lane & 15;
    const int half = lane >> 4;

    v8f acc00 = {}, acc01 = {}, acc10 = {}, acc11 = {};

    const unsigned short* arow0 = ctx + (size_t)(m0 + l16) * EDIM + half * 8;
    const unsigned short* arow1 = arow0 + (size_t)16 * EDIM;
    const unsigned short* brow0 = W + (size_t)(n0 + l16) * EDIM + half * 8;
    const unsigned short* brow1 = brow0 + (size_t)16 * EDIM;

    for (int k0 = 0; k0 < EDIM; k0 += 32) {
        __builtin_prefetch(brow0 + k0 + 128, 0, 1);
        BF16Frag a0, a1, b0, b1;
        a0.q[0] = *(const u32x4*)(arow0 + k0);
        a0.q[1] = *(const u32x4*)(arow0 + k0 + 16);
        a1.q[0] = *(const u32x4*)(arow1 + k0);
        a1.q[1] = *(const u32x4*)(arow1 + k0 + 16);
        b0.q[0] = *(const u32x4*)(brow0 + k0);
        b0.q[1] = *(const u32x4*)(brow0 + k0 + 16);
        b1.q[0] = *(const u32x4*)(brow1 + k0);
        b1.q[1] = *(const u32x4*)(brow1 + k0 + 16);
        acc00 = wmma_bf16(a0.v, b0.v, acc00);
        acc01 = wmma_bf16(a0.v, b1.v, acc01);
        acc10 = wmma_bf16(a1.v, b0.v, acc10);
        acc11 = wmma_bf16(a1.v, b1.v, acc11);
    }

#pragma unroll
    for (int tile = 0; tile < 4; ++tile) {
        const v8f& c = (tile == 0) ? acc00 : (tile == 1) ? acc01 : (tile == 2) ? acc10 : acc11;
        const int mb = (tile >> 1) << 4;
        const int nb = (tile & 1) << 4;
#pragma unroll
        for (int r = 0; r < 8; ++r) {
            const int row = m0 + mb + r + half * 8;
            const int col = n0 + nb + l16;
            out[(size_t)row * EDIM + col] = c[r] + bias[col];
        }
    }
}

// ---------------------------------------------------------------------------
extern "C" void kernel_launch(void* const* d_in, const int* in_sizes, int n_in,
                              void* d_out, int out_size, void* d_ws, size_t ws_size,
                              hipStream_t stream) {
    (void)in_sizes; (void)n_in; (void)out_size; (void)ws_size;
    const float* x     = (const float*)d_in[0];
    const float* qw    = (const float*)d_in[1];
    const float* kw    = (const float*)d_in[2];
    const float* vw    = (const float*)d_in[3];
    const float* ow    = (const float*)d_in[4];
    const float* qbias = (const float*)d_in[5];
    const float* kbias = (const float*)d_in[6];
    const float* vbias = (const float*)d_in[7];
    const float* obias = (const float*)d_in[8];

    // Workspace layout (u16 units). Every byte read is written earlier in the
    // same call (no cross-call state). Total = 92.3 MB.
    const size_t MAT  = (size_t)MDIM * EDIM;   // 8192x1024
    const size_t WMAT = (size_t)EDIM * EDIM;   // 1024x1024
    unsigned short* xb  = (unsigned short*)d_ws;
    unsigned short* qb  = xb + MAT;
    unsigned short* kb  = qb + MAT;
    unsigned short* vT  = kb + MAT;            // [B,H,D,T]
    unsigned short* ctx = vT + MAT;
    unsigned short* wqb = ctx + MAT;
    unsigned short* wkb = wqb + WMAT;
    unsigned short* wvb = wkb + WMAT;
    unsigned short* wob = wvb + WMAT;
    float* out = (float*)d_out;

    // Stage 0: one-shot bf16 conversion (x: 8.4M elems, weights: 1M each).
    cvt_f32_to_bf16<<<(int)(MAT / (256 * 8)), 256, 0, stream>>>(x, xb);
    cvt_f32_to_bf16<<<(int)(WMAT / (256 * 8)), 256, 0, stream>>>(qw, wqb);
    cvt_f32_to_bf16<<<(int)(WMAT / (256 * 8)), 256, 0, stream>>>(kw, wkb);
    cvt_f32_to_bf16<<<(int)(WMAT / (256 * 8)), 256, 0, stream>>>(vw, wvb);
    cvt_f32_to_bf16<<<(int)(WMAT / (256 * 8)), 256, 0, stream>>>(ow, wob);

    // Stage 1: QKV projection. 3 matrices x 8192 tiles, 8 waves/block.
    qkv_gemm_bf16<<<3072, 256, 0, stream>>>(xb, wqb, wkb, wvb,
                                            qbias, kbias, vbias, qb, kb, vT);
    // Stage 2: flash attention. 64 (b,h) x 128 query tiles = 8192 waves.
    attn_flash_bf16<<<1024, 256, 0, stream>>>(qb, kb, vT, ctx);
    // Stage 3: output projection. 8192 tiles.
    out_proj_bf16<<<1024, 256, 0, stream>>>(ctx, wob, obias, out);
}
